// RelativePositionTransformerLayer_979252543789
// MI455X (gfx1250) — matrix operand order repaired and checked by reading, not compiled, not run on hardware
//
#include <hip/hip_runtime.h>
#include <math.h>

// ---------------------------------------------------------------------------
// MI455X (gfx1250) transformer layer: bf16 WMMA GEMMs + flash attention.
// B=4 S=1024 D=1024 H=16 HD=64 F=4096.  ~120 GFLOP matmul work -> compute
// bound on the matrix pipe; everything routed through v_wmma_f32_16x16x32_bf16
// with f32 accumulation, f32 residual/LayerNorm path.
// GEMM LDS staging uses GLOBAL_LOAD_ASYNC_TO_LDS_B128 (ASYNCcnt) with double
// buffering so tile fill overlaps WMMA issue; attention prefetches the next
// key/value chunk with global_prefetch.
// ---------------------------------------------------------------------------

typedef __bf16 bf16;
typedef __attribute__((ext_vector_type(16))) __bf16 v16bf;
typedef __attribute__((ext_vector_type(8)))  __bf16 v8bf;
typedef __attribute__((ext_vector_type(8)))  float  v8f;

#define B_  4
#define S_  1024
#define D_  1024
#define H_  16
#define HD_ 64
#define F_  4096

// ---------------- WMMA helpers (layouts per CDNA5 ISA §7.12.2) ---

__device__ __forceinline__ v8f wmma_bf16(v16bf a, v16bf b, v8f c) {
  return __builtin_amdgcn_wmma_f32_16x16x32_bf16(
      /*neg_a=*/false, a, /*neg_b=*/false, b,
      /*c_mod=*/(short)0, c, /*reuse_a=*/false, /*reuse_b=*/false);
}

// A-fragment 16x32 (MxK): lane L holds row M=L%16; hh=L/16 selects K halves:
// elements e=0..7 -> K = hh*8 + e ; e=8..15 -> K = 16 + hh*8 + (e-8).
__device__ __forceinline__ v16bf frag_a(const bf16* p, int hh) {
  v8bf lo = *(const v8bf*)(p + hh * 8);
  v8bf hi = *(const v8bf*)(p + 16 + hh * 8);
  v16bf r;
#pragma unroll
  for (int i = 0; i < 8; ++i) { r[i] = lo[i]; r[i + 8] = hi[i]; }
  return r;
}

// B-fragment 32x16 (KxN) loaded from BT[n][k] storage: lane L holds column
// N=L%16; elements e=0..7 -> K = hh*8 + e ; e=8..15 -> K = 16 + hh*8 + (e-8).
__device__ __forceinline__ v16bf frag_b(const bf16* p, int hh) {
  v8bf lo = *(const v8bf*)(p + hh * 8);
  v8bf hi = *(const v8bf*)(p + 16 + hh * 8);
  v16bf r;
#pragma unroll
  for (int i = 0; i < 8; ++i) { r[i] = lo[i]; r[i + 8] = hi[i]; }
  return r;
}

__device__ __forceinline__ v8f zero8() {
  v8f z = {0.f, 0.f, 0.f, 0.f, 0.f, 0.f, 0.f, 0.f};
  return z;
}

// ---- async global->LDS copy (16 bytes / lane), tracked by ASYNCcnt ---------
// Generic shared pointers carry the LDS byte offset in their low 32 bits
// (ISA 10.2: LDS aperture address low bits are the LDS offset).
__device__ __forceinline__ void async_cp16(void* lds, const void* g) {
  asm volatile("global_load_async_to_lds_b128 %0, %1, off"
               :
               : "v"((unsigned int)(unsigned long long)(uintptr_t)lds),
                 "v"((unsigned long long)(uintptr_t)g)
               : "memory");
}
__device__ __forceinline__ void wait_async0() {
  asm volatile("s_wait_asynccnt 0" ::: "memory");
}

// ---------------- weight transpose + f32 -> bf16 convert --------------------
// W[K][N] f32  ->  WT[N][K] bf16 (so GEMM B-fragments read contiguous K).
__global__ __launch_bounds__(256) void transpose_cast_k(const float* __restrict__ W,
                                                        bf16* __restrict__ WT,
                                                        int K, int N) {
  __shared__ float t[32][33];
  int tx = threadIdx.x, ty = threadIdx.y;
  int x  = blockIdx.x * 32 + tx;   // N index
  int y0 = blockIdx.y * 32;        // K base
#pragma unroll
  for (int i = ty; i < 32; i += 8)
    t[i][tx] = W[(size_t)(y0 + i) * N + x];
  __syncthreads();
  int xo  = y0 + tx;               // K index in output
  int yo0 = blockIdx.x * 32;       // N base
#pragma unroll
  for (int i = ty; i < 32; i += 8)
    WT[(size_t)(yo0 + i) * K + xo] = (bf16)t[tx][i];
}

// ---------------- LayerNorm (f32 in, bf16 out), one block per row ----------
__global__ __launch_bounds__(256) void layernorm_k(const float* __restrict__ X,
                                                   const float* __restrict__ g,
                                                   const float* __restrict__ bta,
                                                   bf16* __restrict__ out) {
  __shared__ float s1[256], s2[256];
  const int row = blockIdx.x, tid = threadIdx.x;
  const float* xr = X + (size_t)row * D_;
  float v[4], sum = 0.f, sq = 0.f;
#pragma unroll
  for (int i = 0; i < 4; ++i) {
    v[i] = xr[tid + i * 256];
    sum += v[i];
    sq  += v[i] * v[i];
  }
  s1[tid] = sum; s2[tid] = sq;
  __syncthreads();
  for (int o = 128; o > 0; o >>= 1) {
    if (tid < o) { s1[tid] += s1[tid + o]; s2[tid] += s2[tid + o]; }
    __syncthreads();
  }
  const float mu   = s1[0] * (1.f / D_);
  const float var  = s2[0] * (1.f / D_) - mu * mu;
  const float rstd = rsqrtf(var + 1e-5f);
  bf16* orow = out + (size_t)row * D_;
#pragma unroll
  for (int i = 0; i < 4; ++i) {
    int c = tid + i * 256;
    orow[c] = (bf16)((v[i] - mu) * rstd * g[c] + bta[c]);
  }
}

// ---------------- generic bf16 GEMM: C = A[MxK] * BT[NxK]^T ------------------
// 128 threads = 4 waves; 64x64 block tile; each wave 32x32 (2x2 WMMA tiles).
// Double-buffered LDS filled by async global->LDS copies; one barrier per
// K-step; WMMAs overlap the next tile's async fill.
__global__ __launch_bounds__(128) void gemm_bf16_k(const bf16* __restrict__ A,
                                                   const bf16* __restrict__ BT,
                                                   const float* __restrict__ bias,
                                                   const float* __restrict__ resid,
                                                   float* __restrict__ outF,
                                                   bf16* __restrict__ outH,
                                                   int M, int N, int K, int doGelu) {
  __shared__ __align__(16) bf16 As[2][64 * 32];
  __shared__ __align__(16) bf16 Bs[2][64 * 32];
  const int tid = threadIdx.x;
  const int n0 = blockIdx.x * 64, m0 = blockIdx.y * 64;
  const int wid = tid >> 5, lane = tid & 31, lr = lane & 15, hh = lane >> 4;
  const int wm = wid >> 1, wn = wid & 1;
  const int lrow = tid >> 1, lpart = tid & 1;

  const bf16* gArow = A  + (size_t)(m0 + lrow) * K + lpart * 16;
  const bf16* gBrow = BT + (size_t)(n0 + lrow) * K + lpart * 16;

  v8f acc[2][2];
#pragma unroll
  for (int i = 0; i < 2; ++i)
#pragma unroll
    for (int j = 0; j < 2; ++j) acc[i][j] = zero8();

  // issue async fill of a tile buffer for K-offset k0
  auto issue = [&](int buf, int k0) {
    bf16* la = &As[buf][lrow * 32 + lpart * 16];
    bf16* lb = &Bs[buf][lrow * 32 + lpart * 16];
    async_cp16(la,     gArow + k0);
    async_cp16(la + 8, gArow + k0 + 8);
    async_cp16(lb,     gBrow + k0);
    async_cp16(lb + 8, gBrow + k0 + 8);
  };

  issue(0, 0);
  wait_async0();
  __syncthreads();

  int buf = 0;
  for (int k0 = 0; k0 < K; k0 += 32) {
    if (k0 + 32 < K) issue(buf ^ 1, k0 + 32);   // overlaps with WMMAs below

    v16bf af[2], bfr[2];
    af[0]  = frag_a(&As[buf][(wm * 32 + lr) * 32], hh);
    af[1]  = frag_a(&As[buf][(wm * 32 + 16 + lr) * 32], hh);
    bfr[0] = frag_b(&Bs[buf][(wn * 32 + lr) * 32], hh);
    bfr[1] = frag_b(&Bs[buf][(wn * 32 + 16 + lr) * 32], hh);
#pragma unroll
    for (int mi = 0; mi < 2; ++mi)
#pragma unroll
      for (int ni = 0; ni < 2; ++ni)
        acc[mi][ni] = wmma_bf16(af[mi], bfr[ni], acc[mi][ni]);

    wait_async0();       // next buffer's copies (this wave) complete
    __syncthreads();     // visible to all waves; cur reads already consumed
    buf ^= 1;
  }

#pragma unroll
  for (int mi = 0; mi < 2; ++mi) {
#pragma unroll
    for (int ni = 0; ni < 2; ++ni) {
      const int gc = n0 + wn * 32 + ni * 16 + lr;
      const float bs = bias[gc];
#pragma unroll
      for (int r = 0; r < 8; ++r) {
        const int gr = m0 + wm * 32 + mi * 16 + r + 8 * hh;
        float vv = acc[mi][ni][r] + bs;
        if (doGelu) vv = 0.5f * vv * (1.f + erff(vv * 0.70710678118654752f));
        if (resid)  vv += resid[(size_t)gr * N + gc];
        if (outF)   outF[(size_t)gr * N + gc] = vv;
        if (outH)   outH[(size_t)gr * N + gc] = (bf16)vv;
      }
    }
  }
}

// ---------------- flash attention: one wave per (b, head, 16-query tile) ----
__global__ __launch_bounds__(128) void attn_k(const bf16* __restrict__ Q,
                                              const bf16* __restrict__ Km,
                                              const bf16* __restrict__ Vm,
                                              const unsigned char* __restrict__ mask,
                                              bf16* __restrict__ Out) {
  __shared__ __align__(16) bf16 Pl[4 * 16 * 32];   // P staging (per wave)
  __shared__ __align__(16) bf16 Vt[4 * 64 * 32];   // V^T tile   (per wave)
  const int tid = threadIdx.x, wid = tid >> 5, lane = tid & 31;
  const int lr = lane & 15, hh = lane >> 4;
  const int t = blockIdx.x * 4 + wid;              // 0..4095 tasks
  const int b = t >> 10, hidx = (t >> 6) & 15, q0 = (t & 63) * 16;

  const bf16* Qb = Q  + (size_t)b * S_ * D_ + hidx * HD_;
  const bf16* Kb = Km + (size_t)b * S_ * D_ + hidx * HD_;
  const bf16* Vb = Vm + (size_t)b * S_ * D_ + hidx * HD_;
  const unsigned char* mb = mask + (size_t)b * S_;
  bf16* Pw = &Pl[wid * 16 * 32];
  bf16* Vw = &Vt[wid * 64 * 32];

  // Q fragments (K = head dim 64 -> two 32-wide A fragments)
  const v16bf aq0 = frag_a(Qb + (size_t)(q0 + lr) * D_, hh);
  const v16bf aq1 = frag_a(Qb + (size_t)(q0 + lr) * D_ + 32, hh);

  v8f o[4];
#pragma unroll
  for (int i = 0; i < 4; ++i) o[i] = zero8();
  float m[8], l[8];
#pragma unroll
  for (int r = 0; r < 8; ++r) { m[r] = -3.0e38f; l[r] = 0.f; }

  for (int kc = 0; kc < S_; kc += 32) {
    // prefetch next chunk's K/V rows toward the WGP while this chunk computes
    if (kc + 32 < S_) {
      __builtin_prefetch(Kb + (size_t)(kc + 32 + lane) * D_, 0, 0);
      __builtin_prefetch(Vb + (size_t)(kc + 32 + lane) * D_, 0, 0);
    }

    // ---- stage V chunk transposed into LDS: Vw[hd][key], key contiguous ----
    {
      const bf16* vr = Vb + (size_t)(kc + lane) * D_;
      bf16 vbuf[64];
#pragma unroll
      for (int c = 0; c < 8; ++c) {
        v8bf x = *(const v8bf*)(vr + c * 8);
#pragma unroll
        for (int i = 0; i < 8; ++i) vbuf[c * 8 + i] = x[i];
      }
#pragma unroll
      for (int j = 0; j < 64; ++j) Vw[j * 32 + lane] = vbuf[j];
    }

    // ---- scores: S = Q @ K^T for 32 keys (two 16-wide N tiles) ----
    v8f s[2];
#pragma unroll
    for (int nt = 0; nt < 2; ++nt) {
      const bf16* kr = Kb + (size_t)(kc + nt * 16 + lr) * D_;
      v16bf bk0 = frag_b(kr, hh);
      v16bf bk1 = frag_b(kr + 32, hh);
      v8f z = zero8();
      z = wmma_bf16(aq0, bk0, z);
      z = wmma_bf16(aq1, bk1, z);
      s[nt] = z;
    }

    const bool mk0 = mb[kc + lr] != 0;
    const bool mk1 = mb[kc + 16 + lr] != 0;
    const int k0i = kc + lr, k1i = kc + 16 + lr;

    // ---- scale + rel-bias + mask; online softmax stats per row ----
    float cm[8];
#pragma unroll
    for (int r = 0; r < 8; ++r) {
      const int qi = q0 + r + 8 * hh;
      float s0 = s[0][r] * 0.125f - 0.1f * fabsf((float)(qi - k0i));
      float s1 = s[1][r] * 0.125f - 0.1f * fabsf((float)(qi - k1i));
      if (mk0) s0 = -1.0e30f;
      if (mk1) s1 = -1.0e30f;
      s[0][r] = s0; s[1][r] = s1;
      cm[r] = fmaxf(s0, s1);
    }
#pragma unroll
    for (int r = 0; r < 8; ++r) {
      cm[r] = fmaxf(cm[r], __shfl_xor(cm[r], 1, 32));
      cm[r] = fmaxf(cm[r], __shfl_xor(cm[r], 2, 32));
      cm[r] = fmaxf(cm[r], __shfl_xor(cm[r], 4, 32));
      cm[r] = fmaxf(cm[r], __shfl_xor(cm[r], 8, 32));
    }
#pragma unroll
    for (int r = 0; r < 8; ++r) {
      const float mn = fmaxf(m[r], cm[r]);
      const float al = __expf(m[r] - mn);
      const float p0 = __expf(s[0][r] - mn);
      const float p1 = __expf(s[1][r] - mn);
      s[0][r] = p0; s[1][r] = p1;
      float rs = p0 + p1;
      rs += __shfl_xor(rs, 1, 32);
      rs += __shfl_xor(rs, 2, 32);
      rs += __shfl_xor(rs, 4, 32);
      rs += __shfl_xor(rs, 8, 32);
      l[r] = l[r] * al + rs;
      m[r] = mn;
#pragma unroll
      for (int nt2 = 0; nt2 < 4; ++nt2) o[nt2][r] *= al;
    }

    // ---- C-layout P -> LDS -> A-layout fragment ----
#pragma unroll
    for (int r = 0; r < 8; ++r) {
      Pw[(r + 8 * hh) * 32 + lr]      = (bf16)s[0][r];
      Pw[(r + 8 * hh) * 32 + 16 + lr] = (bf16)s[1][r];
    }
    asm volatile("s_wait_dscnt 0" ::: "memory");  // same-wave LDS RAW

    const v16bf ap = frag_a(&Pw[lr * 32], hh);
#pragma unroll
    for (int nt2 = 0; nt2 < 4; ++nt2) {
      v16bf bv = frag_b(&Vw[(nt2 * 16 + lr) * 32], hh);
      o[nt2] = wmma_bf16(ap, bv, o[nt2]);
    }
  }

  // ---- normalize and store ----
#pragma unroll
  for (int r = 0; r < 8; ++r) {
    const float inv = 1.f / l[r];
    const size_t rowoff = ((size_t)b * S_ + q0 + r + 8 * hh) * D_ + hidx * HD_;
#pragma unroll
    for (int nt2 = 0; nt2 < 4; ++nt2)
      Out[rowoff + nt2 * 16 + lr] = (bf16)(o[nt2][r] * inv);
  }
}

// ---------------------------------------------------------------------------
extern "C" void kernel_launch(void* const* d_in, const int* in_sizes, int n_in,
                              void* d_out, int out_size, void* d_ws, size_t ws_size,
                              hipStream_t stream) {
  (void)in_sizes; (void)n_in; (void)out_size; (void)ws_size;
  const float*         src  = (const float*)d_in[0];
  const unsigned char* mask = (const unsigned char*)d_in[1];
  const float* Wq = (const float*)d_in[2];  const float* bq = (const float*)d_in[3];
  const float* Wk = (const float*)d_in[4];  const float* bk = (const float*)d_in[5];
  const float* Wv = (const float*)d_in[6];  const float* bv = (const float*)d_in[7];
  const float* Wo = (const float*)d_in[8];  const float* bo = (const float*)d_in[9];
  const float* W1 = (const float*)d_in[10]; const float* b1 = (const float*)d_in[11];
  const float* W2 = (const float*)d_in[12]; const float* b2 = (const float*)d_in[13];
  const float* g1 = (const float*)d_in[14]; const float* be1 = (const float*)d_in[15];
  const float* g2 = (const float*)d_in[16]; const float* be2 = (const float*)d_in[17];

  char* w = (char*)d_ws;
  size_t off = 0;
  auto carve = [&](size_t bytes) -> char* {
    char* p = w + off;
    off += (bytes + 255) & ~(size_t)255;
    return p;
  };
  const size_t NT = (size_t)B_ * S_;  // 4096 tokens
  bf16*  WqT = (bf16*)carve((size_t)D_ * D_ * 2);
  bf16*  WkT = (bf16*)carve((size_t)D_ * D_ * 2);
  bf16*  WvT = (bf16*)carve((size_t)D_ * D_ * 2);
  bf16*  WoT = (bf16*)carve((size_t)D_ * D_ * 2);
  bf16*  W1T = (bf16*)carve((size_t)D_ * F_ * 2);
  bf16*  W2T = (bf16*)carve((size_t)F_ * D_ * 2);
  bf16*  hb  = (bf16*)carve(NT * D_ * 2);
  bf16*  qb  = (bf16*)carve(NT * D_ * 2);
  bf16*  kb  = (bf16*)carve(NT * D_ * 2);
  bf16*  vb  = (bf16*)carve(NT * D_ * 2);
  bf16*  ao  = (bf16*)carve(NT * D_ * 2);
  float* xf  = (float*)carve(NT * D_ * 4);
  bf16*  ff  = (bf16*)carve(NT * F_ * 2);

  const dim3 tb(32, 8);
  transpose_cast_k<<<dim3(D_ / 32, D_ / 32), tb, 0, stream>>>(Wq, WqT, D_, D_);
  transpose_cast_k<<<dim3(D_ / 32, D_ / 32), tb, 0, stream>>>(Wk, WkT, D_, D_);
  transpose_cast_k<<<dim3(D_ / 32, D_ / 32), tb, 0, stream>>>(Wv, WvT, D_, D_);
  transpose_cast_k<<<dim3(D_ / 32, D_ / 32), tb, 0, stream>>>(Wo, WoT, D_, D_);
  transpose_cast_k<<<dim3(F_ / 32, D_ / 32), tb, 0, stream>>>(W1, W1T, D_, F_);
  transpose_cast_k<<<dim3(D_ / 32, F_ / 32), tb, 0, stream>>>(W2, W2T, F_, D_);

  // h = LN1(src)
  layernorm_k<<<(int)NT, 256, 0, stream>>>(src, g1, be1, hb);

  // Q/K/V projections
  const dim3 gDD(D_ / 64, (int)NT / 64);
  gemm_bf16_k<<<gDD, 128, 0, stream>>>(hb, WqT, bq, nullptr, nullptr, qb,
                                       (int)NT, D_, D_, 0);
  gemm_bf16_k<<<gDD, 128, 0, stream>>>(hb, WkT, bk, nullptr, nullptr, kb,
                                       (int)NT, D_, D_, 0);
  gemm_bf16_k<<<gDD, 128, 0, stream>>>(hb, WvT, bv, nullptr, nullptr, vb,
                                       (int)NT, D_, D_, 0);

  // attention (flash, online softmax, rel-bias + padding mask)
  attn_k<<<(B_ * H_ * (S_ / 16)) / 4, 128, 0, stream>>>(qb, kb, vb, mask, ao);

  // x = src + attn @ Wo + bo   (f32)
  gemm_bf16_k<<<gDD, 128, 0, stream>>>(ao, WoT, bo, src, xf, nullptr,
                                       (int)NT, D_, D_, 0);

  // h2 = LN2(x)
  layernorm_k<<<(int)NT, 256, 0, stream>>>(xf, g2, be2, hb);

  // ff = gelu(h2 @ W1 + b1)
  gemm_bf16_k<<<dim3(F_ / 64, (int)NT / 64), 128, 0, stream>>>(
      hb, W1T, b1, nullptr, nullptr, ff, (int)NT, F_, D_, 1);

  // out = x + ff @ W2 + b2   (f32, final output)
  gemm_bf16_k<<<dim3(D_ / 64, (int)NT / 64), 128, 0, stream>>>(
      ff, W2T, b2, xf, (float*)d_out, nullptr, (int)NT, D_, F_, 0);
}
